// MultiHeadAttention_76321568850236
// MI455X (gfx1250) — matrix operand order repaired and checked by reading, not compiled, not run on hardware
//
#include <hip/hip_runtime.h>

typedef __bf16 bf16_t;
typedef __attribute__((ext_vector_type(16))) __bf16 v16bf;
typedef __attribute__((ext_vector_type(8)))  float  v8f;

#ifndef __has_builtin
#define __has_builtin(x) 0
#endif
#if __has_builtin(__builtin_amdgcn_global_load_async_to_lds_b32)
#define HAVE_ASYNC_CP 1
#endif

typedef __attribute__((address_space(1))) int g1int_t;  // global
typedef __attribute__((address_space(3))) int l3int_t;  // LDS

union FragBF {
    v16bf v;
    unsigned int u[8];
    unsigned short h[16];
};

__device__ __forceinline__ v8f wmma_bf16(v16bf a, v16bf b, v8f c) {
    return __builtin_amdgcn_wmma_f32_16x16x32_bf16(
        /*neg_a=*/false, a, /*neg_b=*/false, b,
        /*c_mod=*/(short)0, c, /*reuse_a=*/false, /*reuse_b=*/false);
}

// One dword memory -> LDS copy via the gfx1250 async path (ASYNCcnt).
__device__ __forceinline__ void cp_b32(void* lds_dst, const void* gsrc) {
#if defined(HAVE_ASYNC_CP)
    __builtin_amdgcn_global_load_async_to_lds_b32(
        (g1int_t*)gsrc, (l3int_t*)lds_dst, 0, 0);
#else
    *(unsigned int*)lds_dst = *(const unsigned int*)gsrc;
#endif
}

__device__ __forceinline__ void wait_cp() {
#if defined(HAVE_ASYNC_CP)
#if __has_builtin(__builtin_amdgcn_s_wait_asynccnt)
    __builtin_amdgcn_s_wait_asynccnt(0);
#else
    asm volatile("s_wait_asynccnt 0x0" ::: "memory");
#endif
#endif
}

// A fragment: 16x32 bf16 tile, element (m,k) at base[m*ld + k].
// lane L -> row = L%16; half = L/16;
// u[0..3]: K = half*8 + 2j,2j+1 ; u[4..7]: K = 16 + half*8 + 2j,2j+1
__device__ __forceinline__ v16bf load_fragA(const bf16_t* base, int ld) {
    FragBF f;
    const int lane = threadIdx.x & 31;
    const int row  = lane & 15;
    const int half = lane >> 4;
    const bf16_t* p = base + row * ld;
#pragma unroll
    for (int j = 0; j < 4; ++j) {
        f.u[j]     = *(const unsigned int*)(p + half * 8 + 2 * j);
        f.u[4 + j] = *(const unsigned int*)(p + 16 + half * 8 + 2 * j);
    }
    return f.v;
}

// B fragment (32x16, element (k,n)) from K-CONTIGUOUS storage:
// element (k,n) at base[n*ld + k]; lane L -> n = L%16;
// u[0..3]: K = (L/16)*8 + 2j ; u[4..7]: K = 16 + (L/16)*8 + 2j. All dword loads.
__device__ __forceinline__ v16bf load_fragB_T(const bf16_t* base, int ld) {
    FragBF f;
    const int lane = threadIdx.x & 31;
    const int n  = lane & 15;
    const int kb = (lane >> 4) * 8;
    const bf16_t* p = base + n * ld + kb;
#pragma unroll
    for (int j = 0; j < 4; ++j) {
        f.u[j]     = *(const unsigned int*)(p + 2 * j);
        f.u[4 + j] = *(const unsigned int*)(p + 16 + 2 * j);
    }
    return f.v;
}

// ---------------------------------------------------------------------------
// f32 -> bf16 streaming conversion (8 elements / thread)
// ---------------------------------------------------------------------------
__global__ __launch_bounds__(256)
void cvt_kernel(const float* __restrict__ src, bf16_t* __restrict__ dst) {
    size_t i = ((size_t)blockIdx.x * 256 + threadIdx.x) * 8;
    float4 a = *(const float4*)(src + i);
    float4 b = *(const float4*)(src + i + 4);
    union { bf16_t h[8]; uint4 u; } o;
    o.h[0] = (bf16_t)a.x; o.h[1] = (bf16_t)a.y;
    o.h[2] = (bf16_t)a.z; o.h[3] = (bf16_t)a.w;
    o.h[4] = (bf16_t)b.x; o.h[5] = (bf16_t)b.y;
    o.h[6] = (bf16_t)b.z; o.h[7] = (bf16_t)b.w;
    *(uint4*)(dst + i) = o.u;
}

// ---------------------------------------------------------------------------
// f32 -> bf16 + transpose for weights: dst[n*1024 + k] = (bf16)src[k*1024 + n]
// 64x64 tiles through LDS; reads and writes both coalesced.
// ---------------------------------------------------------------------------
__global__ __launch_bounds__(256)
void cvtT_kernel(const float* __restrict__ src, bf16_t* __restrict__ dst) {
    __shared__ bf16_t T[64][66];
    const int tm = (blockIdx.x / 16) * 64;  // k-tile
    const int tn = (blockIdx.x % 16) * 64;  // n-tile
    const int tid = threadIdx.x;
#pragma unroll
    for (int i = 0; i < 16; ++i) {
        int idx = tid + i * 256;
        int r = idx >> 6, c = idx & 63;     // r = k, c = n
        T[r][c] = (bf16_t)src[(size_t)(tm + r) * 1024 + tn + c];
    }
    __syncthreads();
#pragma unroll
    for (int i = 0; i < 16; ++i) {
        int idx = tid + i * 256;
        int n = idx >> 6, k = idx & 63;
        dst[(size_t)(tn + n) * 1024 + tm + k] = T[k][n];
    }
}

#define BM 128
#define BN 64
#define BK 64
#define LDA 66   // BK + 2 padding (even -> dword aligned)
#define LDB 66   // BK + 2 (B tile stored [n][k], k contiguous)

// ---------------------------------------------------------------------------
// Projection: X[4096,1024](bf16) @ Wt[n][k](bf16) + b, scaled, -> bf16
//   vt==0: head layout [B*H, S, 64]
//   vt==1: transposed head layout [B*H, 64, S]  (for V)
// Double-buffered LDS, async memory->LDS staging, all-dword fragment loads.
// ---------------------------------------------------------------------------
__global__ __launch_bounds__(256)
void proj_kernel(const bf16_t* __restrict__ X, const bf16_t* __restrict__ Wt,
                 const float* __restrict__ bias, bf16_t* __restrict__ Out,
                 float out_scale, int vt) {
    __shared__ bf16_t As[2][BM * LDA];
    __shared__ bf16_t Bs[2][BN * LDB];
    const int tilesN = 1024 / BN; // 16
    const int tileM = (blockIdx.x / tilesN) * BM;
    const int tileN = (blockIdx.x % tilesN) * BN;
    const int tid = threadIdx.x;
    const int w = tid >> 5, wm = w >> 1, wn = w & 1;

    v8f acc[2][2] = {};

    auto stage = [&](int buf, int kt) {
#pragma unroll
        for (int i = 0; i < 16; ++i) {              // 128x64 A tile (dwords)
            int idx = tid + i * 256;
            int r = idx >> 5, c2 = idx & 31;        // 32 dwords per row
            cp_b32(&As[buf][r * LDA + 2 * c2],
                   &X[(size_t)(tileM + r) * 1024 + kt + 2 * c2]);
        }
#pragma unroll
        for (int i = 0; i < 8; ++i) {               // 64(n) x 64(k) B tile (dwords)
            int idx = tid + i * 256;
            int r = idx >> 5, c2 = idx & 31;        // r = n, c2 = k-dword
            cp_b32(&Bs[buf][r * LDB + 2 * c2],
                   &Wt[(size_t)(tileN + r) * 1024 + kt + 2 * c2]);
        }
    };

    stage(0, 0);
    const int T = 1024 / BK; // 16
    for (int t = 0; t < T; ++t) {
        wait_cp();
        __syncthreads();
        if (t + 1 < T) stage((t + 1) & 1, (t + 1) * BK);
        const bf16_t* Ab = As[t & 1];
        const bf16_t* Bb = Bs[t & 1];
#pragma unroll
        for (int ks = 0; ks < BK; ks += 32) {
            v16bf a0 = load_fragA(Ab + (wm * 32) * LDA + ks, LDA);
            v16bf a1 = load_fragA(Ab + (wm * 32 + 16) * LDA + ks, LDA);
            v16bf b0 = load_fragB_T(Bb + (wn * 32) * LDB + ks, LDB);
            v16bf b1 = load_fragB_T(Bb + (wn * 32 + 16) * LDB + ks, LDB);
            acc[0][0] = wmma_bf16(a0, b0, acc[0][0]);
            acc[0][1] = wmma_bf16(a0, b1, acc[0][1]);
            acc[1][0] = wmma_bf16(a1, b0, acc[1][0]);
            acc[1][1] = wmma_bf16(a1, b1, acc[1][1]);
        }
        __syncthreads();
    }

    const int lane = tid & 31;
    const int nloc = lane & 15;
    const int half = lane >> 4;
#pragma unroll
    for (int sm = 0; sm < 2; ++sm)
#pragma unroll
        for (int sn = 0; sn < 2; ++sn) {
            int col = tileN + wn * 32 + sn * 16 + nloc;
            float bv = bias[col];
            int h = col >> 6, dk = col & 63;
#pragma unroll
            for (int i = 0; i < 8; ++i) {
                int row = tileM + wm * 32 + sm * 16 + i + 8 * half;
                int b = row >> 11, s = row & 2047;
                bf16_t val = (bf16_t)((acc[sm][sn][i] + bv) * out_scale);
                if (vt)
                    Out[((size_t)(b * 16 + h) * 64 + dk) * 2048 + s] = val;
                else
                    Out[(((size_t)(b * 16 + h) * 2048) + s) * 64 + dk] = val;
            }
        }
}

// ---------------------------------------------------------------------------
// Output projection: concat[4096,1024](bf16) @ WoT[n][k](bf16) + bo -> f32
// ---------------------------------------------------------------------------
__global__ __launch_bounds__(256)
void oproj_kernel(const bf16_t* __restrict__ X, const bf16_t* __restrict__ Wt,
                  const float* __restrict__ bias, float* __restrict__ Out) {
    __shared__ bf16_t As[2][BM * LDA];
    __shared__ bf16_t Bs[2][BN * LDB];
    const int tilesN = 1024 / BN;
    const int tileM = (blockIdx.x / tilesN) * BM;
    const int tileN = (blockIdx.x % tilesN) * BN;
    const int tid = threadIdx.x;
    const int w = tid >> 5, wm = w >> 1, wn = w & 1;

    v8f acc[2][2] = {};

    auto stage = [&](int buf, int kt) {
#pragma unroll
        for (int i = 0; i < 16; ++i) {
            int idx = tid + i * 256;
            int r = idx >> 5, c2 = idx & 31;
            cp_b32(&As[buf][r * LDA + 2 * c2],
                   &X[(size_t)(tileM + r) * 1024 + kt + 2 * c2]);
        }
#pragma unroll
        for (int i = 0; i < 8; ++i) {
            int idx = tid + i * 256;
            int r = idx >> 5, c2 = idx & 31;
            cp_b32(&Bs[buf][r * LDB + 2 * c2],
                   &Wt[(size_t)(tileN + r) * 1024 + kt + 2 * c2]);
        }
    };

    stage(0, 0);
    const int T = 1024 / BK;
    for (int t = 0; t < T; ++t) {
        wait_cp();
        __syncthreads();
        if (t + 1 < T) stage((t + 1) & 1, (t + 1) * BK);
        const bf16_t* Ab = As[t & 1];
        const bf16_t* Bb = Bs[t & 1];
#pragma unroll
        for (int ks = 0; ks < BK; ks += 32) {
            v16bf a0 = load_fragA(Ab + (wm * 32) * LDA + ks, LDA);
            v16bf a1 = load_fragA(Ab + (wm * 32 + 16) * LDA + ks, LDA);
            v16bf b0 = load_fragB_T(Bb + (wn * 32) * LDB + ks, LDB);
            v16bf b1 = load_fragB_T(Bb + (wn * 32 + 16) * LDB + ks, LDB);
            acc[0][0] = wmma_bf16(a0, b0, acc[0][0]);
            acc[0][1] = wmma_bf16(a0, b1, acc[0][1]);
            acc[1][0] = wmma_bf16(a1, b0, acc[1][0]);
            acc[1][1] = wmma_bf16(a1, b1, acc[1][1]);
        }
        __syncthreads();
    }

    const int lane = tid & 31;
    const int nloc = lane & 15;
    const int half = lane >> 4;
#pragma unroll
    for (int sm = 0; sm < 2; ++sm)
#pragma unroll
        for (int sn = 0; sn < 2; ++sn) {
            int col = tileN + wn * 32 + sn * 16 + nloc;
            float bv = bias[col];
#pragma unroll
            for (int i = 0; i < 8; ++i) {
                int row = tileM + wm * 32 + sm * 16 + i + 8 * half;
                Out[(size_t)row * 1024 + col] = acc[sm][sn][i] + bv;
            }
        }
}

// ---------------------------------------------------------------------------
// Flash attention, causal. Block = one (b,h) x 128 query rows; 8 waves,
// each wave owns 16 query rows. K tiles [key][dk], V tiles [dk][key]
// (V pre-transposed by its projection) -> all fragment loads are dwords.
// K/V double-buffered via async copies. Scale folded into Q.
// ---------------------------------------------------------------------------
#define LDK 66   // K tile row (64 dk + 2)
#define LDV 34   // V tile row (32 keys + 2)
__global__ __launch_bounds__(256)
void attn_kernel(const bf16_t* __restrict__ Qh, const bf16_t* __restrict__ Kh,
                 const bf16_t* __restrict__ Vt, bf16_t* __restrict__ Cc) {
    __shared__ bf16_t Ks[2][32 * LDK];
    __shared__ bf16_t Vs[2][64 * LDV];
    __shared__ bf16_t Ps[8][16 * 34];

    const int nqb = 2048 / 128; // 16
    const int bh = blockIdx.x / nqb;
    const int qb = (blockIdx.x % nqb) * 128;
    const int tid = threadIdx.x, w = tid >> 5, lane = tid & 31;
    const int qw = qb + w * 16;
    const int half = lane >> 4, nloc = lane & 15;

    const bf16_t* Qp = Qh + (size_t)bh * 2048 * 64;
    const bf16_t* Kp = Kh + (size_t)bh * 2048 * 64;
    const bf16_t* Vp = Vt + (size_t)bh * 64 * 2048;  // [dk][s]

    // Q fragments (16 rows x 64 dk) held in VGPRs for the whole kernel
    v16bf qa[2];
    {
        FragBF f0, f1;
        const bf16_t* p = Qp + (size_t)(qw + (lane & 15)) * 64;
#pragma unroll
        for (int j = 0; j < 4; ++j) {
            f0.u[j]     = *(const unsigned int*)(p + half * 8 + 2 * j);
            f0.u[4 + j] = *(const unsigned int*)(p + 16 + half * 8 + 2 * j);
            f1.u[j]     = *(const unsigned int*)(p + 32 + half * 8 + 2 * j);
            f1.u[4 + j] = *(const unsigned int*)(p + 48 + half * 8 + 2 * j);
        }
        qa[0] = f0.v; qa[1] = f1.v;
    }

    v8f o[4] = {};
    float mrow[8], lrow[8];
#pragma unroll
    for (int i = 0; i < 8; ++i) { mrow[i] = -1e30f; lrow[i] = 0.f; }

    auto stageKV = [&](int buf, int kb) {
#pragma unroll
        for (int i = 0; i < 4; ++i) {               // K: 64 keys x 32 dk (1024 dwords)
            int idx = tid + i * 256;
            int kr = idx >> 4, c2 = idx & 15;       // 16 dwords per key row
            cp_b32(&Ks[buf][kr * LDK + 2 * c2], &Kp[(size_t)(kb + kr) * 64 + 2 * c2]);
        }
#pragma unroll
        for (int i = 0; i < 4; ++i) {               // V: 64 dk x 32 keys (1024 dwords)
            int idx = tid + i * 256;
            int r = idx >> 4, c2 = idx & 15;        // r = dk, 16 dwords per dk row
            cp_b32(&Vs[buf][r * LDV + 2 * c2], &Vp[(size_t)r * 2048 + kb + 2 * c2]);
        }
    };

    const int nkt = (qb + 128) / 32; // causal: key tiles needed by this block
    stageKV(0, 0);
    for (int t = 0; t < nkt; ++t) {
        const int kb = t * 32;
        wait_cp();
        __syncthreads();
        if (t + 1 < nkt) stageKV((t + 1) & 1, kb + 32);
        const bf16_t* Kb = Ks[t & 1];
        const bf16_t* Vb = Vs[t & 1];

        // scores: Q(16x64) x K^T(64x32) -> two 16x16 f32 tiles
        v8f c0 = {}, c1 = {};
        v16bf bk00 = load_fragB_T(Kb, LDK);                 // keys 0-15,  dk 0-31
        v16bf bk01 = load_fragB_T(Kb + 32, LDK);            // keys 0-15,  dk 32-63
        v16bf bk10 = load_fragB_T(Kb + 16 * LDK, LDK);      // keys 16-31, dk 0-31
        v16bf bk11 = load_fragB_T(Kb + 16 * LDK + 32, LDK); // keys 16-31, dk 32-63
        c0 = wmma_bf16(qa[0], bk00, c0);
        c0 = wmma_bf16(qa[1], bk01, c0);
        c1 = wmma_bf16(qa[0], bk10, c1);
        c1 = wmma_bf16(qa[1], bk11, c1);

        // causal mask + online softmax (row m lives in vgpr m%8 of half m/8)
        float mnew[8];
#pragma unroll
        for (int i = 0; i < 8; ++i) {
            int q = qw + i + 8 * half;
            float s0 = c0[i], s1 = c1[i];
            if (kb + nloc > q)      s0 = -1e30f;
            if (kb + 16 + nloc > q) s1 = -1e30f;
            c0[i] = s0; c1[i] = s1;
            float t2 = fmaxf(s0, s1);
#pragma unroll
            for (int off = 1; off < 16; off <<= 1)
                t2 = fmaxf(t2, __shfl_xor(t2, off, 32));
            mnew[i] = fmaxf(mrow[i], t2);
        }
#pragma unroll
        for (int i = 0; i < 8; ++i) {
            float alpha = __expf(mrow[i] - mnew[i]);
            float p0 = __expf(c0[i] - mnew[i]);
            float p1 = __expf(c1[i] - mnew[i]);
            float rs = p0 + p1;
#pragma unroll
            for (int off = 1; off < 16; off <<= 1)
                rs += __shfl_xor(rs, off, 32);
            lrow[i] = lrow[i] * alpha + rs;
            mrow[i] = mnew[i];
#pragma unroll
            for (int tt = 0; tt < 4; ++tt) o[tt][i] *= alpha;
            // re-layout P (C-layout -> A-layout) through per-wave LDS scratch
            int prow = i + 8 * half;
            Ps[w][prow * 34 + nloc]      = (bf16_t)p0;
            Ps[w][prow * 34 + 16 + nloc] = (bf16_t)p1;
        }

        // O += P(16x32) x V(32x64); V tile stored [dk][key] -> dword loads
        v16bf pa = load_fragA(&Ps[w][0], 34);
#pragma unroll
        for (int tt = 0; tt < 4; ++tt) {
            v16bf bv = load_fragB_T(Vb + (tt * 16) * LDV, LDV);
            o[tt] = wmma_bf16(pa, bv, o[tt]);
        }
        __syncthreads();
    }

    // finalize: divide by softmax denom, write concat layout
    const int h_ = bh & 15, b_ = bh >> 4;
#pragma unroll
    for (int tt = 0; tt < 4; ++tt)
#pragma unroll
        for (int i = 0; i < 8; ++i) {
            int q = qw + i + 8 * half;
            int dk = tt * 16 + nloc;
            float val = o[tt][i] / lrow[i];
            Cc[((size_t)(b_ * 2048 + q)) * 1024 + h_ * 64 + dk] = (bf16_t)val;
        }
}

// ---------------------------------------------------------------------------
extern "C" void kernel_launch(void* const* d_in, const int* in_sizes, int n_in,
                              void* d_out, int out_size, void* d_ws, size_t ws_size,
                              hipStream_t stream) {
    const float* q  = (const float*)d_in[0];
    const float* k  = (const float*)d_in[1];
    const float* v  = (const float*)d_in[2];
    // d_in[3] = causal mask (computed analytically)
    const float* Wq = (const float*)d_in[4];
    const float* bq = (const float*)d_in[5];
    const float* Wk = (const float*)d_in[6];
    const float* bk = (const float*)d_in[7];
    const float* Wv = (const float*)d_in[8];
    const float* bv = (const float*)d_in[9];
    const float* Wo = (const float*)d_in[10];
    const float* bo = (const float*)d_in[11];

    const size_t ELT_X = (size_t)4096 * 1024; // 4,194,304
    const size_t ELT_W = (size_t)1024 * 1024; // 1,048,576

    bf16_t* p = (bf16_t*)d_ws;
    bf16_t* qbuf = p; p += ELT_X;
    bf16_t* kbuf = p; p += ELT_X;
    bf16_t* vbuf = p; p += ELT_X;
    bf16_t* Wqb  = p; p += ELT_W;   // transposed [n][k]
    bf16_t* Wkb  = p; p += ELT_W;
    bf16_t* Wvb  = p; p += ELT_W;
    bf16_t* Wob  = p; p += ELT_W;
    bf16_t* Qh   = p; p += ELT_X;   // [bh][s][dk]
    bf16_t* Kh   = p; p += ELT_X;   // [bh][s][dk]
    bf16_t* Vh   = p; p += ELT_X;   // [bh][dk][s]  (transposed)
    bf16_t* Cc   = p; p += ELT_X;

    dim3 blk(256);
    cvt_kernel<<<2048, blk, 0, stream>>>(q,  qbuf);
    cvt_kernel<<<2048, blk, 0, stream>>>(k,  kbuf);
    cvt_kernel<<<2048, blk, 0, stream>>>(v,  vbuf);
    cvtT_kernel<<<256, blk, 0, stream>>>(Wq, Wqb);
    cvtT_kernel<<<256, blk, 0, stream>>>(Wk, Wkb);
    cvtT_kernel<<<256, blk, 0, stream>>>(Wv, Wvb);
    cvtT_kernel<<<256, blk, 0, stream>>>(Wo, Wob);

    proj_kernel<<<512, blk, 0, stream>>>(qbuf, Wqb, bq, Qh, 0.125f, 0); // fold 1/sqrt(64)
    proj_kernel<<<512, blk, 0, stream>>>(kbuf, Wkb, bk, Kh, 1.0f, 0);
    proj_kernel<<<512, blk, 0, stream>>>(vbuf, Wvb, bv, Vh, 1.0f, 1);   // V transposed
    attn_kernel<<<512, blk, 0, stream>>>(Qh, Kh, Vh, Cc);
    oproj_kernel<<<512, blk, 0, stream>>>(Cc, Wob, bo, (float*)d_out);
}
